// kNNHyperbolicAttentionLayer_83296595738919
// MI455X (gfx1250) — compile-verified, matrix operand order checked by reading, not run on hardware
//
#include <hip/hip_runtime.h>
#include <math.h>

// ---------------------------------------------------------------------------
// Problem constants (from reference)
// ---------------------------------------------------------------------------
static constexpr int BB   = 2;
static constexpr int NN   = 2048;
static constexpr int DIMC = 1024;
static constexpr int HH   = 16;
static constexpr int KK   = 16;
static constexpr int PDc  = 16;
static constexpr int HD   = 64;      // DIM / H
static constexpr int FF   = 4096;    // 4*DIM
static constexpr int MT   = BB * NN; // 4096 tokens

typedef __attribute__((ext_vector_type(16))) __bf16 v16bf;
typedef __attribute__((ext_vector_type(8)))  __bf16 v8bf;
typedef __attribute__((ext_vector_type(8)))  float  v8f;

// ---------------------------------------------------------------------------
// f32 -> bf16 conversion
// ---------------------------------------------------------------------------
__global__ void cvt_bf16_kernel(const float* __restrict__ src,
                                __bf16* __restrict__ dst, int n) {
    int i = blockIdx.x * 256 + threadIdx.x;
    if (i < n) dst[i] = (__bf16)src[i];
}

// ---------------------------------------------------------------------------
// LayerNorm over last dim (1024), output bf16; one block (256 thr) per row
// ---------------------------------------------------------------------------
__global__ __launch_bounds__(256)
void ln_kernel(const float* __restrict__ x, const float* __restrict__ g,
               const float* __restrict__ b, __bf16* __restrict__ out) {
    const int row = blockIdx.x;
    const int tid = threadIdx.x;
    __shared__ float rs[256];
    __shared__ float rs2[256];

    float v[4];
    float s = 0.f, s2 = 0.f;
#pragma unroll
    for (int i = 0; i < 4; ++i) {
        int e = tid + i * 256;
        float val = x[(size_t)row * DIMC + e];
        v[i] = val;
        s += val;
        s2 += val * val;
    }
    rs[tid] = s; rs2[tid] = s2;
    __syncthreads();
    for (int stp = 128; stp > 0; stp >>= 1) {
        if (tid < stp) { rs[tid] += rs[tid + stp]; rs2[tid] += rs2[tid + stp]; }
        __syncthreads();
    }
    float mean = rs[0] * (1.0f / DIMC);
    float var  = rs2[0] * (1.0f / DIMC) - mean * mean;
    float rinv = rsqrtf(var + 1e-5f);
#pragma unroll
    for (int i = 0; i < 4; ++i) {
        int e = tid + i * 256;
        float nv = (v[i] - mean) * rinv * g[e] + b[e];
        out[(size_t)row * DIMC + e] = (__bf16)nv;
    }
}

// ---------------------------------------------------------------------------
// bf16 WMMA GEMM: C[M,Nd] = A[M,Kd] @ W[Kd,Nd] + bias (+GELU) (+residual).
// Block: 256 threads (8 waves). Block tile 128x128, BK=32, double-buffered.
// Wave (wm,wn) computes rows [wm*32,+32) x cols [wn*64,+64): 8 WMMAs/k-step.
// A tile staged via global_load_async_to_lds_b128 (ASYNCcnt); W tile staged
// transposed (pair-packed ds_store_b32) for b128 fragment loads.
// One barrier per k-step: the top barrier both publishes buffer kb and
// guarantees all waves finished reading the buffer that will be overwritten.
// ---------------------------------------------------------------------------
template <int Kd, int Nd, bool GELU, bool RES, bool OUTF, bool OUTB>
__global__ __launch_bounds__(256)
void gemm_bf16_kernel(const __bf16* __restrict__ A, const __bf16* __restrict__ W,
                      const float* __restrict__ bias,
                      const float* __restrict__ resid,
                      float* __restrict__ outF, __bf16* __restrict__ outB) {
    __shared__ __align__(16) __bf16 Alds[2 * 128 * 32];
    __shared__ __align__(16) __bf16 Wt[2 * 128 * 32];   // Wt[buf][col][k]

    const int tid  = threadIdx.x;
    const int wave = tid >> 5;
    const int lane = tid & 31;
    const int half = lane >> 4;
    const int r    = lane & 15;
    const int wm   = wave >> 1;   // 0..3
    const int wn   = wave & 1;    // 0..1
    const int m0 = blockIdx.y * 128;
    const int n0 = blockIdx.x * 128;

    // staging assignments
    const int rowA = tid >> 1;            // 0..127
    const int colA = (tid & 1) * 16;      // 0 / 16
    const int pK   = tid >> 4;            // k-pair 0..15 (k = 2*pK, 2*pK+1)
    const int n8   = (tid & 15) * 8;      // n-chunk of 8

    const unsigned ldsAbase = (unsigned)(size_t)(void*)&Alds[0];

    auto stage = [&](int k0, int buf) {
        // speculative prefetch of the k-block after this one
        __builtin_prefetch(A + (size_t)(m0 + rowA) * Kd + (k0 + 32) + colA, 0, 1);
        __builtin_prefetch(W + (size_t)(k0 + 32 + 2 * pK) * Nd + n0 + n8, 0, 1);

        // A tile 128x32 row-major via async copy to LDS
        unsigned l0 = ldsAbase + (unsigned)(buf * (128 * 32) + rowA * 32 + colA) * 2u;
        unsigned long long g0 =
            (unsigned long long)(size_t)(A + (size_t)(m0 + rowA) * Kd + k0 + colA);
        asm volatile("global_load_async_to_lds_b128 %0, %1, off"
                     :: "v"(l0), "v"(g0) : "memory");
        asm volatile("global_load_async_to_lds_b128 %0, %1, off offset:16"
                     :: "v"(l0), "v"(g0) : "memory");

        // W tile 32x128 transposed into Wt[buf][128][32], k-pair packed
        union { float4 f; __bf16 h[8]; } u0, u1;
        u0.f = *(const float4*)(W + (size_t)(k0 + 2 * pK) * Nd + n0 + n8);
        u1.f = *(const float4*)(W + (size_t)(k0 + 2 * pK + 1) * Nd + n0 + n8);
#pragma unroll
        for (int e = 0; e < 8; ++e) {
            union { unsigned u; __bf16 h[2]; } pk2;
            pk2.h[0] = u0.h[e];
            pk2.h[1] = u1.h[e];
            *(unsigned*)(&Wt[buf * (128 * 32) + (n8 + e) * 32 + 2 * pK]) = pk2.u;
        }
    };

    v8f c[2][4] = {};

    stage(0, 0);
    int kb = 0;
    for (int k0 = 0; k0 < Kd; k0 += 32, kb ^= 1) {
        asm volatile("s_wait_asynccnt 0x0" ::: "memory");
        __syncthreads();

        // load all fragments from buffer kb
        const __bf16* Ab = Alds + kb * (128 * 32);
        const __bf16* Wb = Wt + kb * (128 * 32);
        v16bf a[2], b[4];
#pragma unroll
        for (int i = 0; i < 2; ++i) {
            const int arow = wm * 32 + i * 16 + r;
            v8bf lo = *(const v8bf*)(Ab + arow * 32 + half * 8);
            v8bf hi = *(const v8bf*)(Ab + arow * 32 + 16 + half * 8);
            a[i] = __builtin_shufflevector(lo, hi, 0, 1, 2, 3, 4, 5, 6, 7,
                                           8, 9, 10, 11, 12, 13, 14, 15);
        }
#pragma unroll
        for (int t = 0; t < 4; ++t) {
            const int bcol = wn * 64 + t * 16 + r;
            v8bf lo = *(const v8bf*)(Wb + bcol * 32 + half * 8);
            v8bf hi = *(const v8bf*)(Wb + bcol * 32 + 16 + half * 8);
            b[t] = __builtin_shufflevector(lo, hi, 0, 1, 2, 3, 4, 5, 6, 7,
                                           8, 9, 10, 11, 12, 13, 14, 15);
        }

        // stage next k-block into the other buffer (overlaps with WMMAs)
        if (k0 + 32 < Kd) stage(k0 + 32, kb ^ 1);

        // issue all WMMAs back-to-back (distinct accumulators: no D->A/B hazard)
#pragma unroll
        for (int i = 0; i < 2; ++i)
#pragma unroll
            for (int t = 0; t < 4; ++t)
                c[i][t] = __builtin_amdgcn_wmma_f32_16x16x32_bf16(
                    false, a[i], false, b[t], (short)0, c[i][t], false, false);
    }

    // --- epilogue: bias (+GELU) (+residual); base + constexpr offsets ---
#pragma unroll
    for (int i = 0; i < 2; ++i) {
#pragma unroll
        for (int t = 0; t < 4; ++t) {
            const int col  = n0 + wn * 64 + t * 16 + r;
            const int row0 = m0 + wm * 32 + i * 16 + half * 8;
            const size_t base = (size_t)row0 * Nd + col;
            const float bv = bias[col];
            const float* rp = resid + base;
            float*       fp = outF + base;
            __bf16*      bp = outB + base;
#pragma unroll
            for (int e = 0; e < 8; ++e) {
                float v = c[i][t][e] + bv;
                if (GELU) v = 0.5f * v * (1.0f + erff(v * 0.70710678118654752f));
                if (RES)  v += rp[(size_t)e * Nd];
                if (OUTF) fp[(size_t)e * Nd] = v;
                if (OUTB) bp[(size_t)e * Nd] = (__bf16)v;
            }
        }
    }
}

// ---------------------------------------------------------------------------
// kNN over Poincare distances: one block (256 threads) per (b,i) token.
// ---------------------------------------------------------------------------
__global__ __launch_bounds__(256)
void knn_kernel(const float* __restrict__ pos, const float* __restrict__ cptr,
                int* __restrict__ idx_out, float* __restrict__ ad_out) {
    const int bid = blockIdx.x;       // b*N + i
    const int b   = bid / NN;
    const int tid = threadIdx.x;

    __shared__ float sdist[NN];
    __shared__ float rv[256];
    __shared__ int   ri[256];
    __shared__ float pis[PDc];

    if (tid < PDc) pis[tid] = pos[(size_t)bid * PDc + tid];
    __syncthreads();

    const float c   = cptr[0];
    const float rsc = rsqrtf(c);
    float xi = 0.f;
#pragma unroll
    for (int p = 0; p < PDc; ++p) xi += pis[p] * pis[p];

    const float* pb = pos + (size_t)b * NN * PDc;
    for (int j = tid; j < NN; j += 256) {
        const float* pj = pb + (size_t)j * PDc;
        float dns = 0.f, xj = 0.f;
#pragma unroll
        for (int p = 0; p < PDc; ++p) {
            float v = pj[p];
            xj += v * v;
            float d = pis[p] - v;
            dns += d * d;
        }
        float den = (1.0f - c * xi) * (1.0f - c * xj) + 1e-8f;
        float arg = 1.0f + 2.0f * c * dns / den;
        arg = fmaxf(arg, 1.0f);
        sdist[j] = acoshf(arg) * rsc;
    }
    __syncthreads();

    for (int k = 0; k < KK; ++k) {
        float bv = 3.0e38f;
        int   bi = 0;
        for (int j = tid; j < NN; j += 256) {
            float d = sdist[j];
            if (d < bv) { bv = d; bi = j; }
        }
        rv[tid] = bv; ri[tid] = bi;
        __syncthreads();
        for (int stp = 128; stp > 0; stp >>= 1) {
            if (tid < stp) {
                if (rv[tid + stp] < rv[tid]) { rv[tid] = rv[tid + stp]; ri[tid] = ri[tid + stp]; }
            }
            __syncthreads();
        }
        if (tid == 0) {
            int bj = ri[0];
            idx_out[(size_t)bid * KK + k] = bj;
            ad_out[(size_t)bid * KK + k]  = rv[0];
            sdist[bj] = 3.0e38f;
        }
        __syncthreads();
    }
}

// ---------------------------------------------------------------------------
// Sparse attention: one block (256 threads) per token (b,n).
// thread = h*16 + kk ; softmax via __shfl_xor within 16-lane groups.
// ---------------------------------------------------------------------------
__global__ __launch_bounds__(256)
void attn_kernel(const float* __restrict__ q, const float* __restrict__ kbuf,
                 const float* __restrict__ vbuf, const int* __restrict__ idx,
                 const float* __restrict__ adist, const float* __restrict__ log_tau,
                 __bf16* __restrict__ ao) {
    const int row = blockIdx.x;       // b*N + n
    const int b   = row / NN;
    const int tid = threadIdx.x;
    const int h   = tid >> 4;
    const int kk  = tid & 15;

    __shared__ float probs[HH * KK];
    __shared__ int   jidx[KK];

    const float tau = expf(log_tau[0]) + 1e-8f;
    const int j = idx[(size_t)row * KK + kk];
    if (h == 0) jidx[kk] = j;

    const float* qp = q    + (size_t)row * DIMC + h * HD;
    const float* kp = kbuf + ((size_t)b * NN + j) * DIMC + h * HD;
    float dot = 0.f;
#pragma unroll
    for (int d = 0; d < HD; d += 4) {
        float4 qa = *(const float4*)(qp + d);
        float4 ka = *(const float4*)(kp + d);
        dot += qa.x * ka.x + qa.y * ka.y + qa.z * ka.z + qa.w * ka.w;
    }
    float s = dot * 0.125f - adist[(size_t)row * KK + kk] / tau;

    float m = s;
#pragma unroll
    for (int off = 8; off >= 1; off >>= 1) m = fmaxf(m, __shfl_xor(m, off));
    float e = expf(s - m);
    float sum = e;
#pragma unroll
    for (int off = 8; off >= 1; off >>= 1) sum += __shfl_xor(sum, off);
    probs[h * KK + kk] = e / sum;
    __syncthreads();

    const int d0 = kk * 4;
    float acc0 = 0.f, acc1 = 0.f, acc2 = 0.f, acc3 = 0.f;
#pragma unroll
    for (int k2 = 0; k2 < KK; ++k2) {
        float p = probs[h * KK + k2];
        const float4 va = *(const float4*)(vbuf + ((size_t)b * NN + jidx[k2]) * DIMC + h * HD + d0);
        acc0 += p * va.x; acc1 += p * va.y; acc2 += p * va.z; acc3 += p * va.w;
    }
    __bf16* op = ao + (size_t)row * DIMC + h * HD + d0;
    op[0] = (__bf16)acc0; op[1] = (__bf16)acc1;
    op[2] = (__bf16)acc2; op[3] = (__bf16)acc3;
}

// ---------------------------------------------------------------------------
// Host launcher
// ---------------------------------------------------------------------------
extern "C" void kernel_launch(void* const* d_in, const int* in_sizes, int n_in,
                              void* d_out, int out_size, void* d_ws, size_t ws_size,
                              hipStream_t stream) {
    const float* x        = (const float*)d_in[0];
    const float* positions= (const float*)d_in[1];
    const float* cptr     = (const float*)d_in[2];
    const float* qw = (const float*)d_in[3];  const float* qb = (const float*)d_in[4];
    const float* kw = (const float*)d_in[5];  const float* kb = (const float*)d_in[6];
    const float* vw = (const float*)d_in[7];  const float* vb = (const float*)d_in[8];
    const float* ow = (const float*)d_in[9];  const float* ob = (const float*)d_in[10];
    const float* w1 = (const float*)d_in[11]; const float* b1 = (const float*)d_in[12];
    const float* w2 = (const float*)d_in[13]; const float* b2 = (const float*)d_in[14];
    const float* g1 = (const float*)d_in[15]; const float* be1= (const float*)d_in[16];
    const float* g2 = (const float*)d_in[17]; const float* be2= (const float*)d_in[18];
    const float* log_tau = (const float*)d_in[19];
    float* out = (float*)d_out;

    // workspace carving (256B aligned)
    char* ws = (char*)d_ws;
    size_t off = 0;
    auto alloc = [&](size_t bytes) -> void* {
        void* p = ws + off;
        off += (bytes + 255) & ~(size_t)255;
        return p;
    };
    __bf16* xn  = (__bf16*)alloc((size_t)MT * DIMC * 2);
    __bf16* qwb = (__bf16*)alloc((size_t)DIMC * DIMC * 2);
    __bf16* kwb = (__bf16*)alloc((size_t)DIMC * DIMC * 2);
    __bf16* vwb = (__bf16*)alloc((size_t)DIMC * DIMC * 2);
    __bf16* owb = (__bf16*)alloc((size_t)DIMC * DIMC * 2);
    __bf16* w1b = (__bf16*)alloc((size_t)DIMC * FF * 2);
    __bf16* w2b = (__bf16*)alloc((size_t)FF * DIMC * 2);
    float*  qf  = (float*)alloc((size_t)MT * DIMC * 4);
    float*  kf  = (float*)alloc((size_t)MT * DIMC * 4);
    float*  vf  = (float*)alloc((size_t)MT * DIMC * 4);
    int*    idxb= (int*)alloc((size_t)MT * KK * 4);
    float*  adb = (float*)alloc((size_t)MT * KK * 4);
    __bf16* aob = (__bf16*)alloc((size_t)MT * DIMC * 2);
    float*  x1  = (float*)alloc((size_t)MT * DIMC * 4);
    __bf16* hb  = (__bf16*)alloc((size_t)MT * DIMC * 2);
    __bf16* gb  = (__bf16*)alloc((size_t)MT * FF * 2);

    // 1. weight conversions
    {
        int n = DIMC * DIMC;
        cvt_bf16_kernel<<<(n + 255) / 256, 256, 0, stream>>>(qw, qwb, n);
        cvt_bf16_kernel<<<(n + 255) / 256, 256, 0, stream>>>(kw, kwb, n);
        cvt_bf16_kernel<<<(n + 255) / 256, 256, 0, stream>>>(vw, vwb, n);
        cvt_bf16_kernel<<<(n + 255) / 256, 256, 0, stream>>>(ow, owb, n);
        int n2 = DIMC * FF;
        cvt_bf16_kernel<<<(n2 + 255) / 256, 256, 0, stream>>>(w1, w1b, n2);
        cvt_bf16_kernel<<<(n2 + 255) / 256, 256, 0, stream>>>(w2, w2b, n2);
    }

    // 2. LN1
    ln_kernel<<<MT, 256, 0, stream>>>(x, g1, be1, xn);

    // 3. Q,K,V projections (WMMA GEMMs): outF only
    {
        dim3 grid(DIMC / 128, MT / 128);
        gemm_bf16_kernel<DIMC, DIMC, false, false, true, false>
            <<<grid, 256, 0, stream>>>(xn, qwb, qb, nullptr, qf, nullptr);
        gemm_bf16_kernel<DIMC, DIMC, false, false, true, false>
            <<<grid, 256, 0, stream>>>(xn, kwb, kb, nullptr, kf, nullptr);
        gemm_bf16_kernel<DIMC, DIMC, false, false, true, false>
            <<<grid, 256, 0, stream>>>(xn, vwb, vb, nullptr, vf, nullptr);
    }

    // 4. kNN top-K
    knn_kernel<<<MT, 256, 0, stream>>>(positions, cptr, idxb, adb);

    // 5. sparse attention -> ao (bf16)
    attn_kernel<<<MT, 256, 0, stream>>>(qf, kf, vf, idxb, adb, log_tau, aob);

    // 6. O projection + residual -> x1 (f32)
    {
        dim3 grid(DIMC / 128, MT / 128);
        gemm_bf16_kernel<DIMC, DIMC, false, true, true, false>
            <<<grid, 256, 0, stream>>>(aob, owb, ob, x, x1, nullptr);
    }

    // 7. LN2
    ln_kernel<<<MT, 256, 0, stream>>>(x1, g2, be2, hb);

    // 8. MLP up + GELU -> gb (bf16 only)
    {
        dim3 grid(FF / 128, MT / 128);
        gemm_bf16_kernel<DIMC, FF, true, false, false, true>
            <<<grid, 256, 0, stream>>>(hb, w1b, b1, nullptr, nullptr, gb);
    }

    // 9. MLP down + residual -> out (f32)
    {
        dim3 grid(DIMC / 128, MT / 128);
        gemm_bf16_kernel<FF, DIMC, false, true, true, false>
            <<<grid, 256, 0, stream>>>(gb, w2b, b2, x1, out, nullptr);
    }
}
